// Self_Attention_29918742184783
// MI455X (gfx1250) — compile-verified, matrix operand order checked by reading, not run on hardware
//
#include <hip/hip_runtime.h>
#include <cstdint>
#include <cstddef>

// Problem constants (match reference)
#define Bb   8
#define Nn   1024
#define Dd   512
#define NHh  8
#define HDd  64          // head dim
#define NORM_FACT 0.044194173824159216f   // 1/sqrt(512)
#define SROW 1028        // padded LDS row stride in floats (== 4 mod 64 banks)

typedef __bf16 bf16_t;
typedef __attribute__((ext_vector_type(16))) __bf16 v16bf;
typedef __attribute__((ext_vector_type(8)))  __bf16 v8bf;
typedef __attribute__((ext_vector_type(8)))  float  v8f;
typedef __attribute__((ext_vector_type(4)))  float  v4f;

// ---------------------------------------------------------------------------
// WMMA fragment loaders.
// 16-bit A-matrix 16x32 layout (ISA 7.12.2): lane<16 -> row=lane, k-chunks
// [kb..kb+7] (VGPR0-3) and [kb+16..kb+23] (VGPR4-7) with kb=0; lane>=16 ->
// row=lane-16, kb=8.  B 32x16 is the same pattern on B^T stored row-major.
// ---------------------------------------------------------------------------
static __device__ __forceinline__
v16bf frag_bf16(const bf16_t* __restrict__ base, int ld, int row_base,
                int k_base, int lane) {
  const int half = lane >> 4;
  const bf16_t* p = base + (size_t)(row_base + (lane & 15)) * ld
                         + k_base + half * 8;
  v8bf lo = *reinterpret_cast<const v8bf*>(p);        // K = k0..k0+7
  v8bf hi = *reinterpret_cast<const v8bf*>(p + 16);   // K = k0+16..k0+23
  return __builtin_shufflevector(lo, hi, 0,1,2,3,4,5,6,7,
                                         8,9,10,11,12,13,14,15);
}

// fp32 source (global or LDS) -> bf16 fragment
static __device__ __forceinline__
v16bf frag_f32_cvt(const float* base, int ld, int row_base,
                   int k_base, int lane) {
  const int half = lane >> 4;
  const float* p = base + (size_t)(row_base + (lane & 15)) * ld
                        + k_base + half * 8;
  v4f a0 = *reinterpret_cast<const v4f*>(p);
  v4f a1 = *reinterpret_cast<const v4f*>(p + 4);
  v4f a2 = *reinterpret_cast<const v4f*>(p + 16);
  v4f a3 = *reinterpret_cast<const v4f*>(p + 20);
  v16bf o;
#pragma unroll
  for (int i = 0; i < 4; ++i) {
    o[i]      = (bf16_t)a0[i];
    o[4 + i]  = (bf16_t)a1[i];
    o[8 + i]  = (bf16_t)a2[i];
    o[12 + i] = (bf16_t)a3[i];
  }
  return o;
}

#define WMMA_BF16(A, Bf, C) \
  __builtin_amdgcn_wmma_f32_16x16x32_bf16(false, (A), false, (Bf), (short)0, (C), false, false)

// ---------------------------------------------------------------------------
// 1) x = normalize(c + u) -> bf16.  One wave per row of 512.
// ---------------------------------------------------------------------------
__global__ void __launch_bounds__(256)
prep_x(const float* __restrict__ c, const float* __restrict__ u,
       bf16_t* __restrict__ xbf) {
  const int row  = blockIdx.x * 8 + (threadIdx.x >> 5);
  const int lane = threadIdx.x & 31;
  const float* pc = c + (size_t)row * Dd + lane * 16;
  const float* pu = u + (size_t)row * Dd + lane * 16;
  float v[16];
  float ss = 0.0f;
#pragma unroll
  for (int i = 0; i < 16; ++i) {
    v[i] = pc[i] + pu[i];
    ss += v[i] * v[i];
  }
#pragma unroll
  for (int off = 16; off > 0; off >>= 1)
    ss += __shfl_xor(ss, off, 32);
  const float inv = 1.0f / fmaxf(sqrtf(ss), 1e-12f);
  bf16_t* px = xbf + (size_t)row * Dd + lane * 16;
#pragma unroll
  for (int i = 0; i < 16; ++i) px[i] = (bf16_t)(v[i] * inv);
}

// ---------------------------------------------------------------------------
// 2) Weights -> bf16, transposed: Wt[sel][n][k] = W[sel][k][n]
// ---------------------------------------------------------------------------
__global__ void __launch_bounds__(256)
prep_w(const float* __restrict__ Wq, const float* __restrict__ Wk,
       const float* __restrict__ Wv, bf16_t* __restrict__ Wt) {
  const int idx = blockIdx.x * 256 + threadIdx.x;       // 3*512*512 total
  const int sel = idx >> 18;
  const int e   = idx & 262143;
  const int n   = e >> 9;
  const int k   = e & 511;
  const float* W = (sel == 0) ? Wq : ((sel == 1) ? Wk : Wv);
  Wt[(size_t)sel * 262144 + (size_t)n * Dd + k] = (bf16_t)W[(size_t)k * Dd + n];
}

// ---------------------------------------------------------------------------
// 3) QKV projection: [8192x512] @ [512x512] x3, one 16x16 tile per wave.
//    Q,K row-major bf16; V stored transposed Vt[b][dglobal][n].
// ---------------------------------------------------------------------------
__global__ void __launch_bounds__(128)
qkv_gemm(const bf16_t* __restrict__ xbf, const bf16_t* __restrict__ Wt,
         const float* __restrict__ bq, const float* __restrict__ bk,
         const float* __restrict__ bv,
         bf16_t* __restrict__ Qbf, bf16_t* __restrict__ Kbf,
         bf16_t* __restrict__ Vt) {
  const int lane = threadIdx.x & 31;
  const int gw   = blockIdx.x * 4 + (threadIdx.x >> 5);  // 0..16383
  const int ct   = gw & 31;                              // col tile (512/16)
  const int rt   = gw >> 5;                              // row tile (8192/16)
  const bf16_t* Wtq = Wt;
  const bf16_t* Wtk = Wt + 262144;
  const bf16_t* Wtv = Wt + 524288;
  v8f cq = {}, ck = {}, cv = {};
  for (int kk = 0; kk < Dd; kk += 32) {
    if (kk + 32 < Dd)
      __builtin_prefetch(xbf + (size_t)(rt * 16) * Dd + kk + 32, 0, 1);
    v16bf a  = frag_bf16(xbf, Dd, rt * 16, kk, lane);
    v16bf wq = frag_bf16(Wtq, Dd, ct * 16, kk, lane);
    v16bf wk = frag_bf16(Wtk, Dd, ct * 16, kk, lane);
    v16bf wv = frag_bf16(Wtv, Dd, ct * 16, kk, lane);
    cq = WMMA_BF16(a, wq, cq);
    ck = WMMA_BF16(a, wk, ck);
    cv = WMMA_BF16(a, wv, cv);
  }
  const int col  = ct * 16 + (lane & 15);
  const int half = lane >> 4;
  const float biasq = bq[col], biask = bk[col], biasv = bv[col];
#pragma unroll
  for (int j = 0; j < 8; ++j) {
    const int r = rt * 16 + j + 8 * half;       // global row = b*1024 + n
    Qbf[(size_t)r * Dd + col] = (bf16_t)(cq[j] + biasq);
    Kbf[(size_t)r * Dd + col] = (bf16_t)(ck[j] + biask);
    const int bidx = r >> 10, n = r & 1023;
    Vt[((size_t)bidx * Dd + col) * Nn + n] = (bf16_t)(cv[j] + biasv);
  }
}

// ---------------------------------------------------------------------------
// 4) Fused attention tail per (b, h, 16-row block):
//    a) S = adj * (Q K^T) * norm_fact      (4 waves x 16 col tiles of WMMA)
//    b) row softmax over the 16x1024 LDS block; write dist (fp32 output)
//    c) att-rows = P @ V from LDS probs (fp32->bf16) x L2-resident Vt
//    dist is written to HBM exactly once and never re-read from HBM.
// ---------------------------------------------------------------------------
__global__ void __launch_bounds__(128)
scores_softmax_pv(const bf16_t* __restrict__ Qbf,
                  const bf16_t* __restrict__ Kbf,
                  const float* __restrict__ adj,
                  const bf16_t* __restrict__ Vt,
                  float* __restrict__ dist, float* __restrict__ att) {
  __shared__ float red[16][8];
  __shared__ float rowstat[16];
  extern __shared__ float S[];                 // 16 * SROW fp32 (padded rows)
  const int lane = threadIdx.x & 31;
  const int wv   = threadIdx.x >> 5;
  const int bid  = blockIdx.x;                 // b*512 + h*64 + rt
  const int rt   = bid & 63;
  const int h    = (bid >> 6) & 7;
  const int b    = bid >> 9;
  const bf16_t* Qh = Qbf + (size_t)b * Nn * Dd + h * HDd;   // ld = 512
  const bf16_t* Kh = Kbf + (size_t)b * Nn * Dd + h * HDd;
  const float* adjb = adj + (size_t)b * Nn * Nn;
  const int rbase = rt * 16;
  const int half  = lane >> 4;

  // --- a) scores into LDS ---
  v16bf a0 = frag_bf16(Qh, Dd, rbase, 0, lane);
  v16bf a1 = frag_bf16(Qh, Dd, rbase, 32, lane);
  for (int t = 0; t < 16; ++t) {
    const int mbase = (wv * 16 + t) * 16;
    v16bf b0 = frag_bf16(Kh, Dd, mbase, 0, lane);
    v16bf b1 = frag_bf16(Kh, Dd, mbase, 32, lane);
    v8f cc = {};
    cc = WMMA_BF16(a0, b0, cc);
    cc = WMMA_BF16(a1, b1, cc);
    const int colm = mbase + (lane & 15);
#pragma unroll
    for (int j = 0; j < 8; ++j) {
      const int mrow = j + 8 * half;                       // 0..15
      const float s = cc[j] * NORM_FACT *
                      adjb[(size_t)(rbase + mrow) * Nn + colm];
      S[mrow * SROW + colm] = s;
    }
  }
  __syncthreads();

  // --- b) softmax: 8 threads per row, 128 cols each; write dist ---
  const int row = threadIdx.x >> 3;
  const int sub = threadIdx.x & 7;
  float* Sr = S + row * SROW + sub * 128;
  float m = -3.402823466e38f;
  for (int i = 0; i < 128; ++i) m = fmaxf(m, Sr[i]);
  red[row][sub] = m;
  __syncthreads();
  if (sub == 0) {
    float mm = red[row][0];
#pragma unroll
    for (int i = 1; i < 8; ++i) mm = fmaxf(mm, red[row][i]);
    rowstat[row] = mm;
  }
  __syncthreads();
  const float rm = rowstat[row];
  float sum = 0.0f;
  for (int i = 0; i < 128; ++i) {
    const float e = __expf(Sr[i] - rm);
    Sr[i] = e;
    sum += e;
  }
  __syncthreads();
  red[row][sub] = sum;
  __syncthreads();
  if (sub == 0) {
    float s = 0.0f;
#pragma unroll
    for (int i = 0; i < 8; ++i) s += red[row][i];
    rowstat[row] = s;
  }
  __syncthreads();
  const float inv = 1.0f / rowstat[row];
  float* drow = dist + (((size_t)(b * NHh + h) * Nn) + rbase + row) * Nn
                     + sub * 128;
  for (int i = 0; i < 128; ++i) {
    const float p = Sr[i] * inv;
    Sr[i] = p;                 // keep normalized probs in LDS for P@V
    drow[i] = p;               // fp32 output (written once)
  }
  __syncthreads();

  // --- c) P @ V for these 16 rows; wave wv owns one 16-col tile of head ---
  const bf16_t* Vh = Vt + (size_t)b * Dd * Nn
                        + (size_t)(h * HDd + wv * 16) * Nn;
  v8f acc = {};
  for (int kk = 0; kk < Nn; kk += 32) {
    v16bf a  = frag_f32_cvt(S, SROW, 0, kk, lane);   // LDS probs -> bf16
    v16bf bb = frag_bf16(Vh, Nn, 0, kk, lane);       // L2-resident V^T
    acc = WMMA_BF16(a, bb, acc);
  }
  const int col = h * HDd + wv * 16 + (lane & 15);
#pragma unroll
  for (int j = 0; j < 8; ++j) {
    const int r = b * Nn + rbase + j + 8 * half;
    att[(size_t)r * Dd + col] = acc[j];
  }
}

// ---------------------------------------------------------------------------
// Launch
// ---------------------------------------------------------------------------
extern "C" void kernel_launch(void* const* d_in, const int* in_sizes, int n_in,
                              void* d_out, int out_size, void* d_ws,
                              size_t ws_size, hipStream_t stream) {
  const float* c   = (const float*)d_in[0];
  const float* u   = (const float*)d_in[1];
  const float* adj = (const float*)d_in[2];
  const float* Wq  = (const float*)d_in[3];
  const float* bq  = (const float*)d_in[4];
  const float* Wk  = (const float*)d_in[5];
  const float* bk  = (const float*)d_in[6];
  const float* Wv  = (const float*)d_in[7];
  const float* bv  = (const float*)d_in[8];

  float* att  = (float*)d_out;                        // [8,1024,512]
  float* dist = att + (size_t)Bb * Nn * Dd;           // [8,8,1024,1024]

  // Workspace layout (bytes), all 16B aligned
  char* ws = (char*)d_ws;
  const size_t XBF_BYTES = (size_t)Bb * Nn * Dd * 2;        // 8 MB
  const size_t WT_BYTES  = (size_t)3 * Dd * Dd * 2;         // 1.5 MB
  const size_t QK_BYTES  = XBF_BYTES;                       // 8 MB each
  bf16_t* xbf = (bf16_t*)ws;
  bf16_t* Wt  = (bf16_t*)(ws + XBF_BYTES);
  bf16_t* Qbf = (bf16_t*)(ws + XBF_BYTES + WT_BYTES);
  bf16_t* Kbf = (bf16_t*)(ws + XBF_BYTES + WT_BYTES + QK_BYTES);
  bf16_t* Vt  = (bf16_t*)(ws + XBF_BYTES + WT_BYTES + 2 * QK_BYTES);

  prep_x<<<Bb * Nn / 8, 256, 0, stream>>>(c, u, xbf);
  prep_w<<<(3 * Dd * Dd) / 256, 256, 0, stream>>>(Wq, Wk, Wv, Wt);
  qkv_gemm<<<4096, 128, 0, stream>>>(xbf, Wt, bq, bk, bv, Qbf, Kbf, Vt);

  const int smem = 16 * SROW * (int)sizeof(float);          // 65,792 B dynamic
  (void)hipFuncSetAttribute((const void*)scores_softmax_pv,
                            hipFuncAttributeMaxDynamicSharedMemorySize, smem);
  scores_softmax_pv<<<Bb * NHh * (Nn / 16), 128, smem, stream>>>(
      Qbf, Kbf, adj, Vt, dist, att);
}
